// CLIPAttention_90331752169834
// MI455X (gfx1250) — matrix operand order repaired and checked by reading, not compiled, not run on hardware
//
#include <hip/hip_runtime.h>

// ---------------------------------------------------------------------------
// CLIP attention (B=2, S=2048, HIDDEN=768, 12 heads x d=64) for gfx1250.
// f16 WMMA (v_wmma_f32_16x16x32_f16) with f32 accumulation, flash-attention
// online softmax so the 24x2048x2048 logits tensor is never materialized.
// ---------------------------------------------------------------------------

typedef _Float16 half16 __attribute__((ext_vector_type(16)));
typedef _Float16 half8  __attribute__((ext_vector_type(8)));
typedef float    float8 __attribute__((ext_vector_type(8)));

#define HIDDEN   768
#define SEQ      2048
#define NHEAD    12
#define HDIM     64
#define ROWS     4096            // B * SEQ
#define SCALE    0.125f          // 1/sqrt(64)

// ---------------------------------------------------------------------------
// A-fragment loader, 16-bit 16x32 A layout (ISA 7.12.2):
//   lanes 0-15 : M = lane,    VGPR0-3 = K 0..7,  VGPR4-7 = K 16..23
//   lanes16-31 : M = lane-16, VGPR0-3 = K 8..15, VGPR4-7 = K 24..31
// p points at element (0,0) of the 16x32 tile; lda in elements.
// ---------------------------------------------------------------------------
__device__ __forceinline__ half16 load_afrag(const _Float16* p, int lda) {
    const int lane = threadIdx.x & 31;
    const int m    = lane & 15;
    const int ks   = (lane >> 4) << 3;          // 0 or 8
    const _Float16* base = p + (size_t)m * lda + ks;
    half8 lo = *(const half8*)(base);
    half8 hi = *(const half8*)(base + 16);
    half16 a;
#pragma unroll
    for (int i = 0; i < 8; ++i) { a[i] = lo[i]; a[i + 8] = hi[i]; }
    return a;
}

// ---------------------------------------------------------------------------
// B-fragment loader, 16-bit 32x16 B layout (ISA 7.12.4 pattern):
//   lanes 0-15 : N = lane,    VGPR0-3 = K 0..7,  VGPR4-7 = K 16..23
//   lanes16-31 : N = lane-16, VGPR0-3 = K 8..15, VGPR4-7 = K 24..31
// Source is B^T stored row-major [N][K]; each lane reads two 16-byte
// runs Bt[n][kb .. kb+7] and Bt[n][kb+16 .. kb+23].
// ---------------------------------------------------------------------------
__device__ __forceinline__ half16 load_bfrag(const _Float16* p, int lda) {
    const int lane = threadIdx.x & 31;
    const int n    = lane & 15;
    const int kb   = (lane >> 4) << 3;          // 0 or 8
    const _Float16* base = p + (size_t)n * lda + kb;
    half8 lo = *(const half8*)(base);
    half8 hi = *(const half8*)(base + 16);
    half16 b;
#pragma unroll
    for (int i = 0; i < 8; ++i) { b[i] = lo[i]; b[i + 8] = hi[i]; }
    return b;
}

__device__ __forceinline__ float8 wmma32(half16 a, half16 b, float8 c) {
    return __builtin_amdgcn_wmma_f32_16x16x32_f16(
        false, a, false, b, (short)0, c, false, false);
}

__device__ __forceinline__ float8 fzero8() {
    float8 z;
#pragma unroll
    for (int i = 0; i < 8; ++i) z[i] = 0.0f;
    return z;
}

// ---------------------------------------------------------------------------
// Kernel 1a: f32 -> f16 copy of x
// ---------------------------------------------------------------------------
__global__ void cvt_f32_f16(const float* __restrict__ src,
                            _Float16* __restrict__ dst, int n) {
    int i = blockIdx.x * blockDim.x + threadIdx.x;
    if (i < n) dst[i] = (_Float16)src[i];
}

// ---------------------------------------------------------------------------
// Kernel 1b: weight transpose + convert: dst[n][k] = (f16) src[k][n]
// ---------------------------------------------------------------------------
__global__ void wtrans(const float* __restrict__ src, _Float16* __restrict__ dst) {
    int i = blockIdx.x * blockDim.x + threadIdx.x;   // over 768*768
    int k = i / HIDDEN;
    int n = i - k * HIDDEN;
    dst[(size_t)n * HIDDEN + k] = (_Float16)src[i];
}

// ---------------------------------------------------------------------------
// Kernel 2: GEMM, 32x32 tile per wave, 2-deep software pipeline.
// out = A[4096x768] * Bt^T + bias, f16 output.
// mode 0: row-major [4096][768] (Q, K)
// mode 1: per-head transposed   Vt[b][h][d][s]  (contiguous half8 store)
// grid (128, 24), 1 wave per block.
// ---------------------------------------------------------------------------
__global__ __launch_bounds__(32)
void gemm_h16(const _Float16* __restrict__ A, const _Float16* __restrict__ Bt,
              const float* __restrict__ bias, _Float16* __restrict__ out,
              int mode) {
    const int row0 = blockIdx.x * 32;
    const int col0 = blockIdx.y * 32;
    const int lane = threadIdx.x & 31;
    const int n    = lane & 15;
    const int mb   = (lane >> 4) << 3;

    float8 acc[2][2];
#pragma unroll
    for (int i = 0; i < 2; ++i)
#pragma unroll
        for (int j = 0; j < 2; ++j) acc[i][j] = fzero8();

    half16 a[2][2], b[2][2];
    // prologue: k-step 0
    a[0][0] = load_afrag(A + (size_t)row0 * HIDDEN, HIDDEN);
    a[0][1] = load_afrag(A + (size_t)(row0 + 16) * HIDDEN, HIDDEN);
    b[0][0] = load_bfrag(Bt + (size_t)col0 * HIDDEN, HIDDEN);
    b[0][1] = load_bfrag(Bt + (size_t)(col0 + 16) * HIDDEN, HIDDEN);

#pragma unroll 2
    for (int k = 0; k < HIDDEN / 32; ++k) {
        const int cur = k & 1, nxt = cur ^ 1;
        if (k + 1 < HIDDEN / 32) {           // prefetch next k-step
            const int k0 = (k + 1) * 32;
            a[nxt][0] = load_afrag(A + (size_t)row0 * HIDDEN + k0, HIDDEN);
            a[nxt][1] = load_afrag(A + (size_t)(row0 + 16) * HIDDEN + k0, HIDDEN);
            b[nxt][0] = load_bfrag(Bt + (size_t)col0 * HIDDEN + k0, HIDDEN);
            b[nxt][1] = load_bfrag(Bt + (size_t)(col0 + 16) * HIDDEN + k0, HIDDEN);
        }
#pragma unroll
        for (int i = 0; i < 2; ++i)
#pragma unroll
            for (int j = 0; j < 2; ++j)
                acc[i][j] = wmma32(a[cur][i], b[cur][j], acc[i][j]);
    }

#pragma unroll
    for (int i = 0; i < 2; ++i)
#pragma unroll
        for (int j = 0; j < 2; ++j) {
            float bb = bias[col0 + j * 16 + n];
#pragma unroll
            for (int r = 0; r < 8; ++r) acc[i][j][r] += bb;
        }

    if (mode == 0) {
#pragma unroll
        for (int i = 0; i < 2; ++i)
#pragma unroll
            for (int j = 0; j < 2; ++j)
#pragma unroll
                for (int r = 0; r < 8; ++r)
                    out[(size_t)(row0 + i * 16 + mb + r) * HIDDEN +
                        col0 + j * 16 + n] = (_Float16)acc[i][j][r];
    } else {
#pragma unroll
        for (int i = 0; i < 2; ++i) {
            const int srow = row0 + i * 16 + mb;     // multiple of 8
            const int bi   = srow >> 11;
            const int ss   = srow & (SEQ - 1);
#pragma unroll
            for (int j = 0; j < 2; ++j) {
                int c  = col0 + j * 16 + n;
                int hh = c >> 6;
                int dd = c & 63;
                half8 v;
#pragma unroll
                for (int r = 0; r < 8; ++r) v[r] = (_Float16)acc[i][j][r];
                *(half8*)(out + ((size_t)(bi * NHEAD + hh) * HDIM + dd) * SEQ + ss) = v;
            }
        }
    }
}

// ---------------------------------------------------------------------------
// Kernel 3: flash attention. grid (128 qtiles, 12 heads, 2 batches), 1 wave.
// Q,K row-major f16 [4096][768]; Vt per-head transposed f16 [b][h][64][2048].
// V fragment loads hoisted above the softmax VALU block to hide latency;
// global_prefetch_b8 for the next K/V tiles.
// ---------------------------------------------------------------------------
__global__ __launch_bounds__(32)
void flash_attn(const _Float16* __restrict__ Q, const _Float16* __restrict__ K,
                const _Float16* __restrict__ Vt, _Float16* __restrict__ attn) {
    __shared__ __align__(16) _Float16 pt[16 * 32];

    const int qt   = blockIdx.x;
    const int h    = blockIdx.y;
    const int bb   = blockIdx.z;
    const int lane = threadIdx.x & 31;
    const int n    = lane & 15;
    const int mb   = (lane >> 4) << 3;

    const _Float16* Qp = Q + (size_t)(bb * SEQ + qt * 16) * HIDDEN + h * HDIM;
    const half16 aq0 = load_afrag(Qp, HIDDEN);       // d = 0..31
    const half16 aq1 = load_afrag(Qp + 32, HIDDEN);  // d = 32..63

    const _Float16* Vh = Vt + (size_t)(bb * NHEAD + h) * HDIM * SEQ;

    float8 o[4];
#pragma unroll
    for (int j = 0; j < 4; ++j) o[j] = fzero8();
    float mrun[8], lrun[8];
#pragma unroll
    for (int r = 0; r < 8; ++r) { mrun[r] = -1e30f; lrun[r] = 0.0f; }

    for (int kt = 0; kt < SEQ / 32; ++kt) {
        const _Float16* Kp =
            K + (size_t)(bb * SEQ + kt * 32) * HIDDEN + h * HDIM;

        // ---- S tile 16x32 = Q(16x64) x K^T : two 16x16 C fragments
        half16 bk00 = load_bfrag(Kp, HIDDEN);                          // keys 0-15,  d 0-31
        half16 bk01 = load_bfrag(Kp + 32, HIDDEN);                     // keys 0-15,  d 32-63
        half16 bk10 = load_bfrag(Kp + (size_t)16 * HIDDEN, HIDDEN);    // keys 16-31, d 0-31
        half16 bk11 = load_bfrag(Kp + (size_t)16 * HIDDEN + 32, HIDDEN);
        float8 s0 = fzero8(), s1 = fzero8();
        s0 = wmma32(aq0, bk00, s0);
        s0 = wmma32(aq1, bk01, s0);
        s1 = wmma32(aq0, bk10, s1);
        s1 = wmma32(aq1, bk11, s1);

        // ---- V fragments: independent of softmax; issue now, use later
        const _Float16* Vp = Vh + kt * 32;
        half16 bv0 = load_bfrag(Vp, SEQ);
        half16 bv1 = load_bfrag(Vp + (size_t)16 * SEQ, SEQ);
        half16 bv2 = load_bfrag(Vp + (size_t)32 * SEQ, SEQ);
        half16 bv3 = load_bfrag(Vp + (size_t)48 * SEQ, SEQ);

        // ---- prefetch next K/V tiles (one line per lane)
        if (kt + 1 < SEQ / 32) {
            __builtin_prefetch((const char*)(Kp + (size_t)32 * HIDDEN) +
                               (size_t)lane * HIDDEN * 2, 0, 3);
            __builtin_prefetch((const char*)(Vp + 32) +
                               (size_t)lane * SEQ * 2, 0, 3);
            __builtin_prefetch((const char*)(Vp + 32) +
                               (size_t)(lane + 32) * SEQ * 2, 0, 3);
        }

        // ---- online softmax (overlaps with V loads in flight)
        float fac[8];
#pragma unroll
        for (int r = 0; r < 8; ++r) {
            float a0 = s0[r] * SCALE, a1 = s1[r] * SCALE;
            float t = fmaxf(a0, a1);                   // row max over keys
            t = fmaxf(t, __shfl_xor(t, 1, 32));
            t = fmaxf(t, __shfl_xor(t, 2, 32));
            t = fmaxf(t, __shfl_xor(t, 4, 32));
            t = fmaxf(t, __shfl_xor(t, 8, 32));
            float nm = fmaxf(mrun[r], t);
            fac[r]  = __expf(mrun[r] - nm);
            mrun[r] = nm;
            float e0 = __expf(a0 - nm), e1 = __expf(a1 - nm);
            s0[r] = e0; s1[r] = e1;
            float su = e0 + e1;                        // row sum
            su += __shfl_xor(su, 1, 32);
            su += __shfl_xor(su, 2, 32);
            su += __shfl_xor(su, 4, 32);
            su += __shfl_xor(su, 8, 32);
            lrun[r] = lrun[r] * fac[r] + su;
        }

#pragma unroll
        for (int j = 0; j < 4; ++j)
#pragma unroll
            for (int r = 0; r < 8; ++r) o[j][r] *= fac[r];

        // ---- C-layout -> A-layout relayout of P through LDS (f16)
#pragma unroll
        for (int r = 0; r < 8; ++r) {
            pt[(mb + r) * 32 + n]      = (_Float16)s0[r];
            pt[(mb + r) * 32 + 16 + n] = (_Float16)s1[r];
        }
        asm volatile("s_wait_dscnt 0" ::: "memory");
        half16 pa = load_afrag((const _Float16*)pt, 32);

        // ---- O += P(16x32) x V(32x64)
        o[0] = wmma32(pa, bv0, o[0]);
        o[1] = wmma32(pa, bv1, o[1]);
        o[2] = wmma32(pa, bv2, o[2]);
        o[3] = wmma32(pa, bv3, o[3]);
        asm volatile("s_wait_dscnt 0" ::: "memory");   // WAR vs next stores
    }

    // normalize and store attention output row-major f16 [4096][768]
#pragma unroll
    for (int r = 0; r < 8; ++r) {
        float inv = 1.0f / lrun[r];
        size_t row = (size_t)(bb * SEQ + qt * 16 + mb + r);
#pragma unroll
        for (int j = 0; j < 4; ++j)
            attn[row * HIDDEN + h * HDIM + j * 16 + n] = (_Float16)(o[j][r] * inv);
    }
}

// ---------------------------------------------------------------------------
// Kernel 4: output projection, 32x32 tile, pipelined, f32 result -> d_out
// ---------------------------------------------------------------------------
__global__ __launch_bounds__(32)
void gemm_f32(const _Float16* __restrict__ A, const _Float16* __restrict__ Bt,
              const float* __restrict__ bias, float* __restrict__ out) {
    const int row0 = blockIdx.x * 32;
    const int col0 = blockIdx.y * 32;
    const int lane = threadIdx.x & 31;
    const int n    = lane & 15;
    const int mb   = (lane >> 4) << 3;

    float8 acc[2][2];
#pragma unroll
    for (int i = 0; i < 2; ++i)
#pragma unroll
        for (int j = 0; j < 2; ++j) acc[i][j] = fzero8();

    half16 a[2][2], b[2][2];
    a[0][0] = load_afrag(A + (size_t)row0 * HIDDEN, HIDDEN);
    a[0][1] = load_afrag(A + (size_t)(row0 + 16) * HIDDEN, HIDDEN);
    b[0][0] = load_bfrag(Bt + (size_t)col0 * HIDDEN, HIDDEN);
    b[0][1] = load_bfrag(Bt + (size_t)(col0 + 16) * HIDDEN, HIDDEN);

#pragma unroll 2
    for (int k = 0; k < HIDDEN / 32; ++k) {
        const int cur = k & 1, nxt = cur ^ 1;
        if (k + 1 < HIDDEN / 32) {
            const int k0 = (k + 1) * 32;
            a[nxt][0] = load_afrag(A + (size_t)row0 * HIDDEN + k0, HIDDEN);
            a[nxt][1] = load_afrag(A + (size_t)(row0 + 16) * HIDDEN + k0, HIDDEN);
            b[nxt][0] = load_bfrag(Bt + (size_t)col0 * HIDDEN + k0, HIDDEN);
            b[nxt][1] = load_bfrag(Bt + (size_t)(col0 + 16) * HIDDEN + k0, HIDDEN);
        }
#pragma unroll
        for (int i = 0; i < 2; ++i)
#pragma unroll
            for (int j = 0; j < 2; ++j)
                acc[i][j] = wmma32(a[cur][i], b[cur][j], acc[i][j]);
    }

#pragma unroll
    for (int i = 0; i < 2; ++i)
#pragma unroll
        for (int j = 0; j < 2; ++j) {
            float bb = bias[col0 + j * 16 + n];
#pragma unroll
            for (int r = 0; r < 8; ++r)
                out[(size_t)(row0 + i * 16 + mb + r) * HIDDEN + col0 + j * 16 + n] =
                    acc[i][j][r] + bb;
        }
}

// ---------------------------------------------------------------------------
// Host-side launcher
// ---------------------------------------------------------------------------
extern "C" void kernel_launch(void* const* d_in, const int* in_sizes, int n_in,
                              void* d_out, int out_size, void* d_ws, size_t ws_size,
                              hipStream_t stream) {
    const float* x  = (const float*)d_in[0];
    const float* Wq = (const float*)d_in[1];
    const float* bq = (const float*)d_in[2];
    const float* Wk = (const float*)d_in[3];
    const float* bk = (const float*)d_in[4];
    const float* Wv = (const float*)d_in[5];
    const float* bv = (const float*)d_in[6];
    const float* Wo = (const float*)d_in[7];
    const float* bo = (const float*)d_in[8];

    char* ws = (char*)d_ws;
    const size_t XB = (size_t)ROWS * HIDDEN * sizeof(_Float16);       // 6.29 MB
    const size_t WB = (size_t)HIDDEN * HIDDEN * sizeof(_Float16);     // 1.18 MB
    _Float16* Xh   = (_Float16*)(ws);
    _Float16* WqT  = (_Float16*)(ws + XB);
    _Float16* WkT  = (_Float16*)(ws + XB + WB);
    _Float16* WvT  = (_Float16*)(ws + XB + 2 * WB);
    _Float16* WoT  = (_Float16*)(ws + XB + 3 * WB);
    _Float16* Qb   = (_Float16*)(ws + XB + 4 * WB);
    _Float16* Kb   = (_Float16*)(ws + XB + 4 * WB + XB);
    _Float16* Vtb  = (_Float16*)(ws + XB + 4 * WB + 2 * XB);
    _Float16* Attn = (_Float16*)(ws + XB + 4 * WB + 3 * XB);

    const int nx = ROWS * HIDDEN;                 // 3,145,728
    const int nw = HIDDEN * HIDDEN;               //   589,824

    cvt_f32_f16<<<nx / 256, 256, 0, stream>>>(x, Xh, nx);
    wtrans<<<nw / 256, 256, 0, stream>>>(Wq, WqT);
    wtrans<<<nw / 256, 256, 0, stream>>>(Wk, WkT);
    wtrans<<<nw / 256, 256, 0, stream>>>(Wv, WvT);
    wtrans<<<nw / 256, 256, 0, stream>>>(Wo, WoT);

    dim3 gg(ROWS / 32, HIDDEN / 32);              // (128, 24)
    gemm_h16<<<gg, 32, 0, stream>>>(Xh, WqT, bq, Qb, 0);
    gemm_h16<<<gg, 32, 0, stream>>>(Xh, WkT, bk, Kb, 0);
    gemm_h16<<<gg, 32, 0, stream>>>(Xh, WvT, bv, Vtb, 1);

    dim3 ga(SEQ / 16, NHEAD, 2);                  // (128, 12, 2)
    flash_attn<<<ga, 32, 0, stream>>>(Qb, Kb, Vtb, Attn);

    gemm_f32<<<gg, 32, 0, stream>>>(Attn, WoT, bo, (float*)d_out);
}
